// ResNetWSL_11235634446957
// MI455X (gfx1250) — compile-verified, matrix operand order checked by reading, not run on hardware
//
#include <hip/hip_runtime.h>
#include <hip/hip_bf16.h>
#include <math.h>

// ---------------------------------------------------------------------------
// ResNetWSL forward for MI455X (gfx1250, wave32, WMMA + TDM).
// Heavy contractions (fc0 conv3x3, conv4 x2) -> v_wmma_f32_16x16x32_f16 with
// B-panels streamed into LDS by the Tensor Data Mover (double-buffered,
// TENSORcnt-tracked). Small tensor algebra -> VALU kernels.
// ---------------------------------------------------------------------------

typedef __attribute__((ext_vector_type(16))) _Float16 v16h;
typedef __attribute__((ext_vector_type(8)))  float    v8f;
typedef __attribute__((ext_vector_type(4)))  unsigned int u32x4;
typedef __attribute__((ext_vector_type(8)))  int      i32x8;
typedef __attribute__((ext_vector_type(4)))  int      i32x4;

#define EPSF 1e-6f

__device__ __forceinline__ v8f wmma_f16(v16h a, v16h b, v8f c) {
  // (neg_a, A, neg_b, B, c_mod, C, reuse_a, reuse_b)
  return __builtin_amdgcn_wmma_f32_16x16x32_f16(false, a, false, b, (short)0, c,
                                                false, false);
}

// 16x32 f16 fragment from LDS (rowstride in halves, rows padded to 40):
// lane<16 -> row=lane, K {0..7,16..23}; lane>=16 -> row=lane-16, K {8..15,24..31}
__device__ __forceinline__ v16h frag_lds(const _Float16* base, int lane, int rowstride) {
  const int row = lane & 15;
  const int ko  = (lane >> 4) << 3;
  const _Float16* q = base + row * rowstride + ko;
  union { v16h v; u32x4 u[2]; } t;
  t.u[0] = *(const u32x4*)(q);
  t.u[1] = *(const u32x4*)(q + 16);
  return t.v;
}

// ---------------------------------------------------------------------------
// TDM: async 2-D tile load (global f16, row-major) -> LDS, rows padded from
// 64B to 80B (pad_interval: 16 DWORDs, pad_amount: 4 DWORDs). TENSORcnt.
// Descriptor packing per CDNA5 ISA.
// 6-arg builtin variant: (u32x4 g0, i32x8 g1, i32x4 g2, i32x4 g3, i32x8, cpol)
// ---------------------------------------------------------------------------
__device__ __forceinline__ void tdm_load_b_panel(unsigned lds_off, const _Float16* gsrc,
                                                 unsigned tensor_d0, unsigned tensor_d1,
                                                 unsigned long row_stride_elems) {
  unsigned long ga = (unsigned long)(size_t)gsrc;
  u32x4 g0;
  g0[0] = 1u;                                          // count=1, user descriptor
  g0[1] = lds_off;                                     // lds_addr (bytes)
  g0[2] = (unsigned)(ga & 0xffffffffu);                // global_addr[31:0]
  g0[3] = (unsigned)((ga >> 32) & 0x01ffffffu)         // global_addr[56:32]
        | (2u << 30);                                  // type = 2 (image)
  i32x8 g1;
  g1[0] = (1 << 16)                                    // data_size = 1 -> 2 bytes
        | (1 << 20)                                    // pad_enable
        | (3 << 22)                                    // pad_interval: 16 DWORDs
        | (3 << 25);                                   // pad_amount:   4 DWORDs
  g1[1] = (int)((tensor_d0 & 0xffffu) << 16);          // tensor_dim0[15:0]
  g1[2] = (int)(((tensor_d0 >> 16) & 0xffffu)          // tensor_dim0[31:16]
        | ((tensor_d1 & 0xffffu) << 16));              // tensor_dim1[15:0]
  g1[3] = (int)(((tensor_d1 >> 16) & 0xffffu)          // tensor_dim1[31:16]
        | (32u << 16));                                // tile_dim0 = 32 elems
  g1[4] = 128;                                         // tile_dim1 = 128 rows (tile_dim2=0)
  g1[5] = (int)(unsigned)(row_stride_elems & 0xffffffffu);  // dim0_stride[31:0]
  g1[6] = (int)(unsigned)((row_stride_elems >> 32) & 0xffffu);
  g1[7] = 0;
  const i32x4 z4 = {0, 0, 0, 0};
  const i32x8 z8 = {0, 0, 0, 0, 0, 0, 0, 0};
  __builtin_amdgcn_tensor_load_to_lds(g0, g1, z4, z4, z8, 0);
}

// ---------------------------------------------------------------------------
// Conversion / packing kernels
// ---------------------------------------------------------------------------
__global__ void k_f32_to_f16(const float* __restrict__ src, _Float16* __restrict__ dst, int n) {
  int i = blockIdx.x * 256 + threadIdx.x;
  if (i < n) dst[i] = (_Float16)src[i];
}

// fc0_w [512][2048][3][3] f32  ->  [512][9*2048] f16 with k = tap*2048 + c
__global__ void k_pack_fc0w(const float* __restrict__ w, _Float16* __restrict__ out) {
  int i = blockIdx.x * 256 + threadIdx.x;           // over 512*2048*9
  if (i >= 512 * 2048 * 9) return;
  int t = i % 9; int rest = i / 9;
  int c = rest % 2048; int o = rest / 2048;
  out[(long)o * 18432 + t * 2048 + c] = (_Float16)w[i];
}

// ---------------------------------------------------------------------------
// fc0: conv3x3 2048->512 as implicit GEMM + BN(eval) + ReLU  (WMMA + TDM)
// grid (4, 7, 64), block 128 (4 waves). Block tile M=32 (spatial), N=128.
// ---------------------------------------------------------------------------
__global__ __launch_bounds__(128)
void k_fc0_conv_wmma(const _Float16* __restrict__ r5h, const _Float16* __restrict__ wfc0,
                     const float* __restrict__ gamma, const float* __restrict__ beta,
                     const float* __restrict__ mean, const float* __restrict__ var,
                     float* __restrict__ y) {
  __shared__ _Float16 atile[32 * 40] __attribute__((aligned(16)));
  __shared__ _Float16 btile[2][128 * 40] __attribute__((aligned(16)));
  const int tid  = threadIdx.x;
  const int lane = tid & 31, wave = tid >> 5;
  const int b    = blockIdx.z;
  const int m0   = blockIdx.y * 32;
  const int n0   = blockIdx.x * 128;     // panel base (global out-channel)
  const int nrw  = wave * 32;            // wave's row offset inside panel
  const v8f vz = {0.f,0.f,0.f,0.f,0.f,0.f,0.f,0.f};
  v8f acc[2][2];
  acc[0][0] = vz; acc[0][1] = vz; acc[1][0] = vz; acc[1][1] = vz;

  const int cc = tid & 31, mseg = tid >> 5;
  const _Float16* src = r5h + (long)b * 2048 * 196;
  const _Float16* wpan = wfc0 + (long)n0 * 18432;

  if (wave == 0)
    tdm_load_b_panel((unsigned)(size_t)&btile[0][0], wpan, 18432u, 512u, 18432ul);

  int cur = 0;
  for (int kg = 0; kg < 18432; kg += 32) {
    const int t  = kg >> 11;             // tap 0..8
    const int c0 = kg & 2047;
    const int dy = t / 3 - 1, dx = t % 3 - 1;
    // stage im2col A tile [32 m][32 k] into LDS
    const _Float16* sc = src + (long)(c0 + cc) * 196;
    __builtin_prefetch(sc + 32 * 196, 0, 1);
    #pragma unroll
    for (int j = 0; j < 8; ++j) {
      int mm = mseg * 8 + j;
      int m  = m0 + mm;
      _Float16 v = (_Float16)0.f;
      if (m < 196) {
        int h = m / 14 + dy, w = m % 14 + dx;
        if ((unsigned)h < 14u && (unsigned)w < 14u) v = sc[h * 14 + w];
      }
      atile[mm * 40 + cc] = v;
    }
    if (wave == 0) {
      if (kg + 32 < 18432) {
        tdm_load_b_panel((unsigned)(size_t)&btile[cur ^ 1][0], wpan + kg + 32,
                         18432u, 512u, 18432ul);
        __builtin_amdgcn_s_wait_tensorcnt((short)1);   // current panel complete
      } else {
        __builtin_amdgcn_s_wait_tensorcnt((short)0);
      }
    }
    __syncthreads();
    v16h a0 = frag_lds(atile, lane, 40);
    v16h a1 = frag_lds(atile + 16 * 40, lane, 40);
    #pragma unroll
    for (int nf = 0; nf < 2; ++nf) {
      v16h bf = frag_lds(&btile[cur][(nrw + nf * 16) * 40], lane, 40);
      acc[0][nf] = wmma_f16(a0, bf, acc[0][nf]);
      acc[1][nf] = wmma_f16(a1, bf, acc[1][nf]);
    }
    __syncthreads();
    cur ^= 1;
  }
  // BN + ReLU epilogue, store y f32 [b][512][196]
  const int ncol = lane & 15;
  const int mofs = (lane >> 4) * 8;
  #pragma unroll
  for (int nf = 0; nf < 2; ++nf) {
    int n = n0 + nrw + nf * 16 + ncol;
    float sc2 = gamma[n] * rsqrtf(var[n] + 1e-5f);
    float mb = mean[n], bb = beta[n];
    #pragma unroll
    for (int mf = 0; mf < 2; ++mf) {
      #pragma unroll
      for (int r = 0; r < 8; ++r) {
        int m = m0 + mf * 16 + mofs + r;
        if (m < 196) {
          float v = (acc[mf][nf][r] - mb) * sc2 + bb;
          y[((long)b * 512 + n) * 196 + m] = fmaxf(v, 0.f);
        }
      }
    }
  }
}

// ---------------------------------------------------------------------------
// EM attention (stage_num=3), per-batch block. Produces x2 (f16) and z_t (f32).
// ---------------------------------------------------------------------------
__global__ __launch_bounds__(256)
void k_em_attention(const float* __restrict__ y, const float* __restrict__ mu0,
                    float* __restrict__ zt, _Float16* __restrict__ x2h) {
  const int b = blockIdx.x, tid = threadIdx.x;
  __shared__ float muL[512 * 8];
  __shared__ float zL[196 * 8];
  __shared__ float sL[8];
  const float* X = y + (long)b * 512 * 196;

  if (tid < 8) {                                  // ||mu0|| over C per class
    float s = 0.f;
    for (int c = 0; c < 512; ++c) { float v = mu0[c * 8 + tid]; s += v * v; }
    sL[tid] = sqrtf(s) + EPSF;
  }
  __syncthreads();
  for (int i = tid; i < 512 * 8; i += 256) muL[i] = mu0[i] / sL[i & 7];
  __syncthreads();

  for (int it = 0; it < 3; ++it) {
    if (tid < 196) {                              // z = softmax_k(x^T mu)
      float d[8] = {0,0,0,0,0,0,0,0};
      for (int c = 0; c < 512; ++c) {
        float xv = X[c * 196 + tid];
        #pragma unroll
        for (int k = 0; k < 8; ++k) d[k] += xv * muL[c * 8 + k];
      }
      float mx = d[0];
      #pragma unroll
      for (int k = 1; k < 8; ++k) mx = fmaxf(mx, d[k]);
      float s = 0.f;
      #pragma unroll
      for (int k = 0; k < 8; ++k) { d[k] = __expf(d[k] - mx); s += d[k]; }
      float inv = 1.f / s;
      #pragma unroll
      for (int k = 0; k < 8; ++k) zL[tid * 8 + k] = d[k] * inv;
    }
    __syncthreads();
    if (tid < 8) {                                // column sums of z
      float s = 0.f;
      for (int m = 0; m < 196; ++m) s += zL[m * 8 + tid];
      sL[tid] = s + EPSF;
    }
    __syncthreads();
    for (int c = tid; c < 512; c += 256) {        // mu = x @ zn
      float a[8] = {0,0,0,0,0,0,0,0};
      for (int m = 0; m < 196; ++m) {
        float xv = X[c * 196 + m];
        #pragma unroll
        for (int k = 0; k < 8; ++k) a[k] += xv * zL[m * 8 + k];
      }
      #pragma unroll
      for (int k = 0; k < 8; ++k) muL[c * 8 + k] = a[k] / sL[k];
    }
    __syncthreads();
    if (tid < 8) {                                // normalize mu over C
      float s = 0.f;
      for (int c = 0; c < 512; ++c) { float v = muL[c * 8 + tid]; s += v * v; }
      sL[tid] = sqrtf(s) + EPSF;
    }
    __syncthreads();
    for (int i = tid; i < 512 * 8; i += 256) muL[i] /= sL[i & 7];
    __syncthreads();
  }
  // x2 = relu(x + mu z^T) as f16 [b][512][196]
  for (int c = tid; c < 512; c += 256) {
    for (int m = 0; m < 196; ++m) {
      float xr = 0.f;
      #pragma unroll
      for (int k = 0; k < 8; ++k) xr += muL[c * 8 + k] * zL[m * 8 + k];
      float v = X[c * 196 + m] + xr;
      x2h[((long)b * 512 + c) * 196 + m] = (_Float16)fmaxf(v, 0.f);
    }
  }
  if (tid < 196) {                                // z_t [b][8][196]
    #pragma unroll
    for (int k = 0; k < 8; ++k) zt[((long)b * 8 + k) * 196 + tid] = zL[tid * 8 + k];
  }
}

// ---------------------------------------------------------------------------
// x1[b, k*512+c, h*14+v] = sum_w x2[b,c,h,w] * z_t[b,k,w,v]   (f16 out)
// ---------------------------------------------------------------------------
__global__ void k_build_x1(const _Float16* __restrict__ x2h, const float* __restrict__ zt,
                           _Float16* __restrict__ x1h, int n) {
  int i = blockIdx.x * 256 + threadIdx.x;
  if (i >= n) return;
  int m = i % 196; int rest = i / 196;
  int c = rest % 512; rest /= 512;
  int k = rest % 8;   int b = rest / 8;
  int h = m / 14, v = m % 14;
  const _Float16* xr = x2h + ((long)b * 512 + c) * 196 + h * 14;
  const float*    zr = zt  + ((long)b * 8   + k) * 196;
  float acc = 0.f;
  #pragma unroll
  for (int w = 0; w < 14; ++w) acc += (float)xr[w] * zr[w * 14 + v];
  x1h[i] = (_Float16)acc;
}

// ---------------------------------------------------------------------------
// conv4 1x1 (4096 -> 2048) as batched WMMA GEMM with TDM-fed B panels.
// MODE 0: A = x1,   epilogue out = (r5 * (acc+bias)) f16   -> xb low half
// MODE 1: A = concat(xblow, r5h), epilogue out = acc+bias   -> xbout
// grid (16, 7, 64), block 128.
// ---------------------------------------------------------------------------
template <int MODE>
__global__ __launch_bounds__(128)
void k_conv4_gemm(const _Float16* __restrict__ srcA0, const _Float16* __restrict__ srcA1,
                  const _Float16* __restrict__ wc4, const float* __restrict__ bias,
                  const float* __restrict__ r5f, _Float16* __restrict__ out) {
  __shared__ _Float16 atile[32 * 40] __attribute__((aligned(16)));
  __shared__ _Float16 btile[2][128 * 40] __attribute__((aligned(16)));
  const int tid  = threadIdx.x;
  const int lane = tid & 31, wave = tid >> 5;
  const int b    = blockIdx.z;
  const int m0   = blockIdx.y * 32;
  const int n0   = blockIdx.x * 128;
  const int nrw  = wave * 32;
  const v8f vz = {0.f,0.f,0.f,0.f,0.f,0.f,0.f,0.f};
  v8f acc[2][2];
  acc[0][0] = vz; acc[0][1] = vz; acc[1][0] = vz; acc[1][1] = vz;
  const int cc = tid & 31, mseg = tid >> 5;
  const _Float16* wpan = wc4 + (long)n0 * 4096;

  if (wave == 0)
    tdm_load_b_panel((unsigned)(size_t)&btile[0][0], wpan, 4096u, 2048u, 4096ul);

  int cur = 0;
  for (int k0 = 0; k0 < 4096; k0 += 32) {
    const int ci = k0 + cc;
    const _Float16* srow;
    if (MODE == 0) {
      srow = srcA0 + ((long)b * 4096 + ci) * 196;
    } else {
      srow = (ci < 2048) ? (srcA0 + ((long)b * 2048 + ci) * 196)
                         : (srcA1 + ((long)b * 2048 + (ci - 2048)) * 196);
    }
    __builtin_prefetch(srow + 32 * 196, 0, 1);
    #pragma unroll
    for (int j = 0; j < 8; ++j) {
      int mm = mseg * 8 + j, m = m0 + mm;
      atile[mm * 40 + cc] = (m < 196) ? srow[m] : (_Float16)0.f;
    }
    if (wave == 0) {
      if (k0 + 32 < 4096) {
        tdm_load_b_panel((unsigned)(size_t)&btile[cur ^ 1][0], wpan + k0 + 32,
                         4096u, 2048u, 4096ul);
        __builtin_amdgcn_s_wait_tensorcnt((short)1);
      } else {
        __builtin_amdgcn_s_wait_tensorcnt((short)0);
      }
    }
    __syncthreads();
    v16h a0 = frag_lds(atile, lane, 40);
    v16h a1 = frag_lds(atile + 16 * 40, lane, 40);
    #pragma unroll
    for (int nf = 0; nf < 2; ++nf) {
      v16h bf = frag_lds(&btile[cur][(nrw + nf * 16) * 40], lane, 40);
      acc[0][nf] = wmma_f16(a0, bf, acc[0][nf]);
      acc[1][nf] = wmma_f16(a1, bf, acc[1][nf]);
    }
    __syncthreads();
    cur ^= 1;
  }
  const int ncol = lane & 15;
  const int mofs = (lane >> 4) * 8;
  #pragma unroll
  for (int nf = 0; nf < 2; ++nf) {
    int n = n0 + nrw + nf * 16 + ncol;
    float bs = bias[n];
    #pragma unroll
    for (int mf = 0; mf < 2; ++mf) {
      #pragma unroll
      for (int r = 0; r < 8; ++r) {
        int m = m0 + mf * 16 + mofs + r;
        if (m < 196) {
          float v = acc[mf][nf][r] + bs;
          long oi = ((long)b * 2048 + n) * 196 + m;
          if (MODE == 0) v *= r5f[oi];
          out[oi] = (_Float16)v;
        }
      }
    }
  }
}

// ---------------------------------------------------------------------------
// down conv (2048->32), GAP, class-wise pools: xg [b][8] and xc [b][196]
// ---------------------------------------------------------------------------
__global__ __launch_bounds__(256)
void k_down_pool(const _Float16* __restrict__ xbout, const float* __restrict__ dw,
                 const float* __restrict__ db, float* __restrict__ xg_out,
                 float* __restrict__ xc) {
  const int b = blockIdx.x, tid = threadIdx.x;
  __shared__ float wch[32 * 128];
  __shared__ float xds[32 * 200];
  __shared__ float gap[32];
  __shared__ float xgs[8];
  float a[32];
  #pragma unroll
  for (int j = 0; j < 32; ++j) a[j] = db[j];
  const int m = tid;
  for (int i0 = 0; i0 < 2048; i0 += 128) {
    for (int t = tid; t < 32 * 128; t += 256) {
      int j = t >> 7, ii = t & 127;
      wch[t] = dw[(long)j * 2048 + i0 + ii];
    }
    __syncthreads();
    if (m < 196) {
      for (int ii = 0; ii < 128; ++ii) {
        float xv = (float)xbout[((long)b * 2048 + i0 + ii) * 196 + m];
        #pragma unroll
        for (int j = 0; j < 32; ++j) a[j] += wch[j * 128 + ii] * xv;
      }
    }
    __syncthreads();
  }
  if (m < 196) {
    #pragma unroll
    for (int j = 0; j < 32; ++j) xds[j * 200 + m] = a[j];
  }
  __syncthreads();
  if (tid < 32) {
    float s = 0.f;
    for (int mm = 0; mm < 196; ++mm) s += xds[tid * 200 + mm];
    gap[tid] = s / 196.f;
  }
  __syncthreads();
  if (tid < 8) {
    float g = 0.25f * (gap[4*tid] + gap[4*tid+1] + gap[4*tid+2] + gap[4*tid+3]);
    xgs[tid] = g;
    xg_out[b * 8 + tid] = g;
  }
  __syncthreads();
  if (m < 196) {
    float s = 0.f;
    #pragma unroll
    for (int g = 0; g < 8; ++g) {
      float p = 0.25f * (xds[(4*g)*200+m] + xds[(4*g+1)*200+m] +
                         xds[(4*g+2)*200+m] + xds[(4*g+3)*200+m]);
      s += p * xgs[g];
    }
    xc[(long)b * 196 + m] = s * 0.125f;
  }
}

// ---------------------------------------------------------------------------
// out2[b] = [mean_m r5 , mean_m xc*r5] @ cls_w.T + cls_b
// ---------------------------------------------------------------------------
__global__ __launch_bounds__(256)
void k_final(const float* __restrict__ r5, const float* __restrict__ xc,
             const float* __restrict__ cw, const float* __restrict__ cb,
             float* __restrict__ out2) {
  const int b = blockIdx.x, tid = threadIdx.x;
  __shared__ float red[256 * 8];
  float a[8] = {0,0,0,0,0,0,0,0};
  const float* xcb = xc + (long)b * 196;
  for (int i = tid; i < 2048; i += 256) {
    const float* row = r5 + ((long)b * 2048 + i) * 196;
    float s1 = 0.f, s2 = 0.f;
    for (int m = 0; m < 196; ++m) { float v = row[m]; s1 += v; s2 += v * xcb[m]; }
    s1 *= (1.f / 196.f); s2 *= (1.f / 196.f);
    #pragma unroll
    for (int t = 0; t < 8; ++t)
      a[t] += cw[(long)t * 4096 + i] * s1 + cw[(long)t * 4096 + 2048 + i] * s2;
  }
  #pragma unroll
  for (int t = 0; t < 8; ++t) red[tid * 8 + t] = a[t];
  __syncthreads();
  for (int s = 128; s > 0; s >>= 1) {
    if (tid < s) {
      #pragma unroll
      for (int t = 0; t < 8; ++t) red[tid * 8 + t] += red[(tid + s) * 8 + t];
    }
    __syncthreads();
  }
  if (tid < 8) out2[b * 8 + tid] = red[tid] + cb[tid];
}

// ---------------------------------------------------------------------------
extern "C" void kernel_launch(void* const* d_in, const int* in_sizes, int n_in,
                              void* d_out, int out_size, void* d_ws, size_t ws_size,
                              hipStream_t stream) {
  (void)in_sizes; (void)n_in; (void)out_size; (void)ws_size;
  const float* r5   = (const float*)d_in[0];
  const float* fc0w = (const float*)d_in[1];
  const float* bng  = (const float*)d_in[2];
  const float* bnb  = (const float*)d_in[3];
  const float* bnm  = (const float*)d_in[4];
  const float* bnv  = (const float*)d_in[5];
  const float* mu0  = (const float*)d_in[6];
  const float* c4w  = (const float*)d_in[7];
  const float* c4b  = (const float*)d_in[8];
  const float* dww  = (const float*)d_in[9];
  const float* dwb  = (const float*)d_in[10];
  const float* clw  = (const float*)d_in[11];
  const float* clb  = (const float*)d_in[12];
  float* out = (float*)d_out;   // [0..511] = xg, [512..1023] = out2

  char* ws = (char*)d_ws;
  size_t off = 0;
  auto alloc = [&](size_t bytes) -> void* {
    void* p = ws + off;
    off = (off + bytes + 255) & ~(size_t)255;
    return p;
  };
  _Float16* r5h   = (_Float16*)alloc((size_t)64 * 2048 * 196 * 2);
  _Float16* wfc0  = (_Float16*)alloc((size_t)512 * 18432 * 2);
  _Float16* wc4   = (_Float16*)alloc((size_t)2048 * 4096 * 2);
  float*    yb    = (float*)   alloc((size_t)64 * 512 * 196 * 4);
  _Float16* x2h   = (_Float16*)alloc((size_t)64 * 512 * 196 * 2);
  float*    zt    = (float*)   alloc((size_t)64 * 8 * 196 * 4);
  _Float16* x1h   = (_Float16*)alloc((size_t)64 * 4096 * 196 * 2);
  _Float16* xblow = (_Float16*)alloc((size_t)64 * 2048 * 196 * 2);
  _Float16* xbout = (_Float16*)alloc((size_t)64 * 2048 * 196 * 2);
  float*    xc    = (float*)   alloc((size_t)64 * 196 * 4);

  int n;
  n = 64 * 2048 * 196;
  k_f32_to_f16<<<(n + 255) / 256, 256, 0, stream>>>(r5, r5h, n);
  n = 512 * 2048 * 9;
  k_pack_fc0w<<<(n + 255) / 256, 256, 0, stream>>>(fc0w, wfc0);
  n = 2048 * 4096;
  k_f32_to_f16<<<(n + 255) / 256, 256, 0, stream>>>(c4w, wc4, n);

  k_fc0_conv_wmma<<<dim3(4, 7, 64), 128, 0, stream>>>(r5h, wfc0, bng, bnb, bnm, bnv, yb);
  k_em_attention<<<64, 256, 0, stream>>>(yb, mu0, zt, x2h);
  n = 64 * 4096 * 196;
  k_build_x1<<<(n + 255) / 256, 256, 0, stream>>>(x2h, zt, x1h, n);

  k_conv4_gemm<0><<<dim3(16, 7, 64), 128, 0, stream>>>(x1h, nullptr, wc4, c4b, r5, xblow);
  k_conv4_gemm<1><<<dim3(16, 7, 64), 128, 0, stream>>>(xblow, r5h, wc4, c4b, nullptr, xbout);

  k_down_pool<<<64, 256, 0, stream>>>(xbout, dww, dwb, out, xc);
  k_final<<<64, 256, 0, stream>>>(r5, xc, clw, clb, out + 512);
}
